// NaiveLinearCrossEntropy_52965536694581
// MI455X (gfx1250) — matrix operand order repaired and checked by reading, not compiled, not run on hardware
//
#include <hip/hip_runtime.h>
#include <hip/hip_bf16.h>

typedef __attribute__((ext_vector_type(16))) __bf16       v16bf;
typedef __attribute__((ext_vector_type(8)))  float        v8f;
typedef __attribute__((ext_vector_type(4)))  unsigned int u32x4;
typedef __attribute__((ext_vector_type(2)))  float        f32x2;
typedef __attribute__((ext_vector_type(2)))  __bf16       bf16x2;

#define NROWS 4096
#define DDIM  2048
#define VOC   50257
#define BM    128
#define BN    128
#define BK    64
#define NK    (DDIM / BK)        // 32
#define ROWTILES (NROWS / BM)    // 32
#define CHUNKS   ((VOC + BN - 1) / BN)  // 393
#define PART     (CHUNKS * 2)    // 786 (two N-half-waves per chunk)
#define LDK   72                 // padded LDS row stride (bf16 elems)

union Frag { v16bf v; u32x4 q[2]; };

__device__ __forceinline__ unsigned pk_bf16(float a, float b) {
    f32x2 f = {a, b};
    bf16x2 h = __builtin_convertvector(f, bf16x2);  // expect v_cvt_pk_bf16_f32
    union { bf16x2 h; unsigned u; } cvt;
    cvt.h = h;
    return cvt.u;
}

__global__ __launch_bounds__(256)
void lce_gemm_partials(const float* __restrict__ Xg, const int* __restrict__ target,
                       const float* __restrict__ Wg, const float* __restrict__ Bg,
                       float* __restrict__ pmax, float* __restrict__ psum,
                       float* __restrict__ tlogit) {
    __shared__ __align__(16) unsigned short Xs[2][BM * LDK];
    __shared__ __align__(16) unsigned short Ws[2][BN * LDK];

    const int tid = threadIdx.x;
    const int rowBase   = blockIdx.x * BM;
    const int chunkBase = blockIdx.y * BN;

    // ---- staging assignment: each thread copies half a row (32 floats) of X and W
    const int srow  = tid >> 1;
    const int shalf = tid & 1;
    const float4* xg = reinterpret_cast<const float4*>(Xg + (size_t)(rowBase + srow) * DDIM + shalf * 32);
    const int wrow  = chunkBase + srow;
    const bool wok  = (wrow < VOC);
    const float4* wg = reinterpret_cast<const float4*>(Wg + (size_t)(wok ? wrow : 0) * DDIM + shalf * 32);

    float4 xv[8], wv[8];
    const float4 z4 = make_float4(0.f, 0.f, 0.f, 0.f);

    auto loadStage = [&](int kt) {
#pragma unroll
        for (int j = 0; j < 8; ++j) {
            xv[j] = xg[kt * 16 + j];
            wv[j] = wok ? wg[kt * 16 + j] : z4;
        }
    };
    auto storeStage = [&](int buf) {
        unsigned xp[16], wp[16];
#pragma unroll
        for (int j = 0; j < 8; ++j) {
            xp[2 * j]     = pk_bf16(xv[j].x, xv[j].y);
            xp[2 * j + 1] = pk_bf16(xv[j].z, xv[j].w);
            wp[2 * j]     = pk_bf16(wv[j].x, wv[j].y);
            wp[2 * j + 1] = pk_bf16(wv[j].z, wv[j].w);
        }
        u32x4* xd = (u32x4*)&Xs[buf][srow * LDK + shalf * 32];
        u32x4* wd = (u32x4*)&Ws[buf][srow * LDK + shalf * 32];
#pragma unroll
        for (int q = 0; q < 4; ++q) {
            u32x4 tx = { xp[4 * q], xp[4 * q + 1], xp[4 * q + 2], xp[4 * q + 3] };
            u32x4 tw = { wp[4 * q], wp[4 * q + 1], wp[4 * q + 2], wp[4 * q + 3] };
            xd[q] = tx;
            wd[q] = tw;
        }
    };

    // ---- wave tiling: 8 waves in 4(M) x 2(N); wave tile 32x64
    const int w     = tid >> 5;
    const int lane  = tid & 31;
    const int half  = lane >> 4;
    const int lc    = lane & 15;
    const int waveM = w & 3;
    const int waveN = w >> 2;

    v8f acc[2][4];
#pragma unroll
    for (int mi = 0; mi < 2; ++mi)
#pragma unroll
        for (int ni = 0; ni < 4; ++ni)
            acc[mi][ni] = (v8f){0.f, 0.f, 0.f, 0.f, 0.f, 0.f, 0.f, 0.f};

    // ---- software pipeline: global->regs, regs->LDS (double-buffered), WMMA
    loadStage(0);
    storeStage(0);
    __syncthreads();
    for (int kt = 0; kt < NK; ++kt) {
        if (kt + 1 < NK) loadStage(kt + 1);     // global loads overlap WMMA
        const int buf = kt & 1;

#pragma unroll
        for (int ks = 0; ks < 2; ++ks) {
            const int kb = ks * 32;
            Frag af[2];
#pragma unroll
            for (int mi = 0; mi < 2; ++mi) {
                const int r0  = waveM * 32 + mi * 16 + lc;
                const int idx = r0 * LDK + kb + half * 8;   // A: K 0..7 / 8..15 then +16
                af[mi].q[0] = *(const u32x4*)&Xs[buf][idx];
                af[mi].q[1] = *(const u32x4*)&Xs[buf][idx + 16];
            }
            Frag bf[4];
#pragma unroll
            for (int ni = 0; ni < 4; ++ni) {
                const int c0  = waveN * 64 + ni * 16 + lc;
                const int idx = c0 * LDK + kb + half * 8;   // B: K 0..7 / 8..15 then +16
                bf[ni].q[0] = *(const u32x4*)&Ws[buf][idx];
                bf[ni].q[1] = *(const u32x4*)&Ws[buf][idx + 16];
            }
#pragma unroll
            for (int mi = 0; mi < 2; ++mi)
#pragma unroll
                for (int ni = 0; ni < 4; ++ni)
                    acc[mi][ni] = __builtin_amdgcn_wmma_f32_16x16x32_bf16(
                        false, af[mi].v, false, bf[ni].v,
                        (short)0, acc[mi][ni], false, false);
        }

        if (kt + 1 < NK) storeStage((kt + 1) & 1);  // fill the other buffer
        __syncthreads();
    }

    // ---- epilogue: bias, per-row online-softmax partials, target logit
    const int colBase = chunkBase + waveN * 64;
#pragma unroll
    for (int ni = 0; ni < 4; ++ni) {
        const int col = colBase + ni * 16 + lc;
        const float bias = (col < VOC) ? Bg[col] : -__builtin_inff();
#pragma unroll
        for (int mi = 0; mi < 2; ++mi) acc[mi][ni] += bias;
    }

    const int part = blockIdx.y * 2 + waveN;
#pragma unroll
    for (int mi = 0; mi < 2; ++mi) {
#pragma unroll
        for (int r = 0; r < 8; ++r) {
            const float v0 = acc[mi][0][r], v1 = acc[mi][1][r];
            const float v2 = acc[mi][2][r], v3 = acc[mi][3][r];
            float mx = fmaxf(fmaxf(v0, v1), fmaxf(v2, v3));
#pragma unroll
            for (int off = 1; off < 16; off <<= 1)
                mx = fmaxf(mx, __shfl_xor(mx, off, 32));
            float se = (mx == -__builtin_inff())
                           ? 0.f
                           : (__expf(v0 - mx) + __expf(v1 - mx) +
                              __expf(v2 - mx) + __expf(v3 - mx));
#pragma unroll
            for (int off = 1; off < 16; off <<= 1)
                se += __shfl_xor(se, off, 32);

            const int row = rowBase + waveM * 32 + mi * 16 + half * 8 + r;
            const int t = target[row];
            float tv = 0.f;
#pragma unroll
            for (int ni = 0; ni < 4; ++ni) {
                const int col = colBase + ni * 16 + lc;
                tv += (col == t) ? acc[mi][ni][r] : 0.f;
            }
#pragma unroll
            for (int off = 1; off < 16; off <<= 1)
                tv += __shfl_xor(tv, off, 32);

            if (lc == 0) {
                pmax[(size_t)part * NROWS + row] = mx;
                psum[(size_t)part * NROWS + row] = se;
                if (t >= colBase && t < colBase + 64) tlogit[row] = tv;
            }
        }
    }
}

__global__ __launch_bounds__(256)
void lce_merge_rows(const float* __restrict__ pmax, const float* __restrict__ psum,
                    const float* __restrict__ tlogit, const int* __restrict__ target,
                    float* __restrict__ nll) {
    const int row = blockIdx.x * blockDim.x + threadIdx.x;
    if (row >= NROWS) return;
    float m = -__builtin_inff(), s = 0.f;
    for (int j = 0; j < PART; ++j) {
        const float pm = pmax[(size_t)j * NROWS + row];
        const float ps = psum[(size_t)j * NROWS + row];
        const float nm = fmaxf(m, pm);
        if (nm == -__builtin_inff()) continue;
        s = s * __expf(m - nm) + ps * __expf(pm - nm);
        m = nm;
    }
    const float logz = m + __logf(s);
    const int t = target[row];
    nll[row] = (t != -100) ? (logz - tlogit[row]) : 0.f;
}

__global__ __launch_bounds__(256)
void lce_final_reduce(const float* __restrict__ nll, const int* __restrict__ target,
                      float* __restrict__ out) {
    __shared__ float ssum[256];
    __shared__ int   scnt[256];
    const int tid = threadIdx.x;
    float s = 0.f; int c = 0;
    for (int i = tid; i < NROWS; i += 256) {
        s += nll[i];
        c += (target[i] != -100) ? 1 : 0;
    }
    ssum[tid] = s; scnt[tid] = c;
    __syncthreads();
    for (int off = 128; off > 0; off >>= 1) {
        if (tid < off) { ssum[tid] += ssum[tid + off]; scnt[tid] += scnt[tid + off]; }
        __syncthreads();
    }
    if (tid == 0) out[0] = ssum[0] / fmaxf((float)scnt[0], 1.f);
}

extern "C" void kernel_launch(void* const* d_in, const int* in_sizes, int n_in,
                              void* d_out, int out_size, void* d_ws, size_t ws_size,
                              hipStream_t stream) {
    (void)in_sizes; (void)n_in; (void)out_size; (void)ws_size;
    const float* x      = (const float*)d_in[0];
    const int*   target = (const int*)d_in[1];
    const float* W      = (const float*)d_in[2];
    const float* b      = (const float*)d_in[3];
    float*       out    = (float*)d_out;

    float* pmax   = (float*)d_ws;                          // PART * NROWS
    float* psum   = pmax + (size_t)PART * NROWS;           // PART * NROWS
    float* tlogit = psum + (size_t)PART * NROWS;           // NROWS
    float* nll    = tlogit + NROWS;                        // NROWS

    dim3 g1(ROWTILES, CHUNKS);
    lce_gemm_partials<<<g1, 256, 0, stream>>>(x, target, W, b, pmax, psum, tlogit);
    lce_merge_rows<<<NROWS / 256, 256, 0, stream>>>(pmax, psum, tlogit, target, nll);
    lce_final_reduce<<<1, 256, 0, stream>>>(nll, target, out);
}
